// AttentionLayer_75299366633967
// MI455X (gfx1250) — matrix operand order repaired and checked by reading, not compiled, not run on hardware
//
#include <hip/hip_runtime.h>
#include <hip/hip_bf16.h>
#include <math.h>

// ---------------------------------------------------------------------------
// Types / helpers for CDNA5 WMMA (wave32): v_wmma_f32_16x16x32_bf16
// ---------------------------------------------------------------------------
typedef __bf16 bf16_t;
typedef __attribute__((ext_vector_type(16))) __bf16 v16bf;
typedef __attribute__((ext_vector_type(8)))  __bf16 v8bf;
typedef __attribute__((ext_vector_type(8)))  float  v8f;

__device__ __forceinline__ bf16_t f2bf(float f) {
    union { float f; unsigned u; } in; in.f = f;
    unsigned u = in.u;
    unsigned r = (u + 0x7FFFu + ((u >> 16) & 1u)) >> 16; // round-to-nearest-even
    union { unsigned short s; bf16_t b; } out;
    out.s = (unsigned short)r;
    return out.b;
}
__device__ __forceinline__ float bf2f(bf16_t b) {
    union { bf16_t b; unsigned short s; } in; in.b = b;
    union { unsigned u; float f; } out; out.u = ((unsigned)in.s) << 16;
    return out.f;
}

// Build a 16-element bf16 fragment from two 16-byte-aligned LDS chunks.
__device__ __forceinline__ v16bf join16(const bf16_t* p0, const bf16_t* p1) {
    union { v16bf v; v8bf h[2]; } u;
    u.h[0] = *(const v8bf*)p0;
    u.h[1] = *(const v8bf*)p1;
    return u.v;
}

__device__ __forceinline__ v8f wmma_bf16(v16bf a, v16bf b, v8f c) {
    return __builtin_amdgcn_wmma_f32_16x16x32_bf16(
        false, a, false, b, (short)0, c, false, false);
}

// ---------------------------------------------------------------------------
// gfx1250 async global->LDS copy (ASYNCcnt), 16B per lane.
// VDST = LDS byte-address VGPR; GVS addressing: SGPR64 base + VGPR32 offset.
// ---------------------------------------------------------------------------
__device__ __forceinline__ void async_copy16(unsigned ldsAddr, unsigned byteOff,
                                             const void* base) {
    asm volatile("global_load_async_to_lds_b128 %0, %1, %2"
                 :: "v"(ldsAddr), "v"(byteOff), "s"(base)
                 : "memory");
}
__device__ __forceinline__ void async_wait_le4() {
    asm volatile("s_wait_asynccnt 0x4" ::: "memory");
}
__device__ __forceinline__ void async_wait_le0() {
    asm volatile("s_wait_asynccnt 0x0" ::: "memory");
}

// ---------------------------------------------------------------------------
// Pre-pass: fp32 -> bf16 flat convert (4 elems / thread)
// ---------------------------------------------------------------------------
__global__ __launch_bounds__(256) void cvt_bf16(const float* __restrict__ in,
                                                bf16_t* __restrict__ out) {
    int i = blockIdx.x * 256 + threadIdx.x;          // float4 index
    float4 v = ((const float4*)in)[i];
    union { bf16_t e[4]; uint2 u; } o;
    o.e[0] = f2bf(v.x); o.e[1] = f2bf(v.y); o.e[2] = f2bf(v.z); o.e[3] = f2bf(v.w);
    ((uint2*)out)[i] = o.u;
}

// ---------------------------------------------------------------------------
// Pre-pass: W[K][N] fp32 -> Wt[N][K] bf16 (tiled transpose)
// Grid (N/32, K/32), block 256 (= 8x32).
// ---------------------------------------------------------------------------
__global__ __launch_bounds__(256) void transpose_cvt_bf16(
    const float* __restrict__ W, bf16_t* __restrict__ Wt, int K, int N) {
    __shared__ float Ts[32][33];
    const int k0 = blockIdx.y * 32, n0 = blockIdx.x * 32;
    const int tx = threadIdx.x & 31, ty = threadIdx.x >> 5;  // ty: 0..7
    #pragma unroll
    for (int i = 0; i < 4; ++i)
        Ts[ty + i * 8][tx] = W[(size_t)(k0 + ty + i * 8) * N + n0 + tx];
    __syncthreads();
    #pragma unroll
    for (int i = 0; i < 4; ++i) {
        int r = ty + i * 8;
        Wt[(size_t)(n0 + r) * K + k0 + tx] = f2bf(Ts[tx][r]);
    }
}

// ---------------------------------------------------------------------------
// GEMM: C[M,N] = act(A[M,K] * W[K,N] + bias); A bf16 row-major, W given as
// Wt[N][K] bf16. Block = 256 threads (8 waves), block tile 128x128, k-step 32,
// double-buffered LDS filled by async global->LDS copies.
// Wave w: rows (w>>1)*32, cols (w&1)*64 -> 2x4 WMMA tiles.
// ---------------------------------------------------------------------------
#define LDT 40   // 32 + 8 bf16 pad, rows stay 16B aligned

__device__ __forceinline__ void stage_panel_async(
    const bf16_t* gbase, bf16_t* lds, int tid, int K, int k0) {
    // 128 rows x 32 cols bf16; 2 async b128 per thread
    #pragma unroll
    for (int i = 0; i < 2; ++i) {
        int idx = tid + i * 256;
        int r = idx >> 2, c8 = (idx & 3) << 3;
        unsigned ldsA = (unsigned)(size_t)(lds + r * LDT + c8);
        unsigned off  = (unsigned)(((unsigned)r * (unsigned)K + (unsigned)(k0 + c8)) * 2u);
        async_copy16(ldsA, off, (const void*)gbase);
    }
}

__global__ __launch_bounds__(256) void gemm_bf16_wmma(
    const bf16_t* __restrict__ Ag, const bf16_t* __restrict__ Wt,
    const float* __restrict__ bias, float* __restrict__ C32,
    bf16_t* __restrict__ C16, int M, int N, int K, int doRelu)
{
    __shared__ alignas(16) bf16_t As[2][128][LDT];
    __shared__ alignas(16) bf16_t Bs[2][128][LDT];   // rows indexed by n

    const int tid  = threadIdx.x;
    const int wave = tid >> 5, lane = tid & 31;
    const int hf   = lane >> 4, l16 = lane & 15;
    const int m0   = blockIdx.y * 128;
    const int n0   = blockIdx.x * 128;
    const int wm   = (wave >> 1) * 32;
    const int wn   = (wave & 1) * 64;

    const bf16_t* Apan = Ag + (size_t)m0 * K;
    const bf16_t* Bpan = Wt + (size_t)n0 * K;

    v8f acc[2][4] = {};

    // Prefetch tile 0
    stage_panel_async(Apan, &As[0][0][0], tid, K, 0);
    stage_panel_async(Bpan, &Bs[0][0][0], tid, K, 0);

    const int nKT = K >> 5;
    for (int kt = 0; kt < nKT; ++kt) {
        __syncthreads();                  // everyone done reading buf (kt+1)&1 (prev use)
        const bool pre = (kt + 1 < nKT);
        if (pre) {
            stage_panel_async(Apan, &As[(kt + 1) & 1][0][0], tid, K, (kt + 1) * 32);
            stage_panel_async(Bpan, &Bs[(kt + 1) & 1][0][0], tid, K, (kt + 1) * 32);
            async_wait_le4();             // tile kt landed; kt+1 may be in flight
        } else {
            async_wait_le0();
        }
        __syncthreads();

        const int b = kt & 1;
        v16bf afrag[2], bfrag[4];
        #pragma unroll
        for (int mt = 0; mt < 2; ++mt) {
            const bf16_t* row = &As[b][wm + mt * 16 + l16][0];
            afrag[mt] = join16(row + hf * 8, row + 16 + hf * 8);
        }
        #pragma unroll
        for (int nt = 0; nt < 4; ++nt) {
            const bf16_t* row = &Bs[b][wn + nt * 16 + l16][0];
            bfrag[nt] = join16(row + hf * 8, row + 16 + hf * 8);
        }
        #pragma unroll
        for (int mt = 0; mt < 2; ++mt)
            #pragma unroll
            for (int nt = 0; nt < 4; ++nt)
                acc[mt][nt] = wmma_bf16(afrag[mt], bfrag[nt], acc[mt][nt]);
    }

    // Epilogue. C layout: vgpr i -> M = i + hf*8, N = l16 (per 16x16 tile).
    #pragma unroll
    for (int mt = 0; mt < 2; ++mt) {
        #pragma unroll
        for (int nt = 0; nt < 4; ++nt) {
            int n = n0 + wn + nt * 16 + l16;
            float bv = bias[n];
            #pragma unroll
            for (int i = 0; i < 8; ++i) {
                int m = m0 + wm + mt * 16 + hf * 8 + i;
                float v = acc[mt][nt][i] + bv;
                if (doRelu) v = fmaxf(v, 0.f);
                if (C16) C16[(size_t)m * N + n] = f2bf(v);
                else     C32[(size_t)m * N + n] = v;
            }
        }
    }
}

// ---------------------------------------------------------------------------
// Causal flash attention. Grid (T/64, H, B), block = 128 threads (4 waves).
// Q/K/V are bf16 [B*T, 1024], head h at cols h*64..h*64+63. Out fp32.
// Per wave: 16 queries; k-tiles of 32 keys; WMMA QK^T and P*V; online softmax.
// K tile staged with async global->LDS copies; V staged transposed.
// ---------------------------------------------------------------------------
__global__ __launch_bounds__(128) void attn_flash_wmma(
    const bf16_t* __restrict__ Qg, const bf16_t* __restrict__ Kg,
    const bf16_t* __restrict__ Vg, float* __restrict__ Og, int Tlen)
{
    __shared__ alignas(16) bf16_t Kb[32][72];     // [key][d]
    __shared__ alignas(16) bf16_t Vt[64][40];     // [d][key]
    __shared__ float  Ss[4][16][33];
    __shared__ alignas(16) bf16_t Pb[4][16][40];
    __shared__ float  mSt[4][16], lSt[4][16], cSt[4][16];

    const int tid  = threadIdx.x;
    const int wave = tid >> 5, lane = tid & 31;
    const int hf   = lane >> 4, l16 = lane & 15;
    const int h    = blockIdx.y, bb = blockIdx.z;
    const int q0b  = blockIdx.x * 64;
    const int q0w  = q0b + wave * 16;

    const size_t base = ((size_t)bb * Tlen) * 1024 + (size_t)h * 64;
    const bf16_t* Kpan = Kg + base;

    // Q fragments (A-layout), scaled by 1/sqrt(64) = 0.125
    v16bf qf[2];
    {
        union { v16bf v; bf16_t e[16]; } u;
        #pragma unroll
        for (int dt = 0; dt < 2; ++dt) {
            #pragma unroll
            for (int e = 0; e < 16; ++e) {
                int d = dt * 32 + ((e < 8 ? e : e + 8) + hf * 8);
                u.e[e] = f2bf(bf2f(Qg[base + (size_t)(q0w + l16) * 1024 + d]) * 0.125f);
            }
            qf[dt] = u.v;
        }
    }

    v8f oacc[4] = {};
    if (lane < 16) { mSt[wave][lane] = -1e30f; lSt[wave][lane] = 0.f; }

    const int nKT = (q0b + 64) / 32;
    for (int kt = 0; kt < nKT; ++kt) {
        const int k0 = kt * 32;
        __syncthreads();
        // Async-stage K tile (32x64 bf16): 2 async b128 per thread
        #pragma unroll
        for (int i = 0; i < 2; ++i) {
            int idx = tid + i * 128;
            int r = idx >> 3, c8 = (idx & 7) << 3;
            unsigned ldsA = (unsigned)(size_t)(&Kb[r][c8]);
            unsigned off  = (unsigned)(((unsigned)(k0 + r) * 1024u + (unsigned)c8) * 2u);
            async_copy16(ldsA, off, (const void*)Kpan);
        }
        // Stage V transposed (sync; values needed in regs)
        #pragma unroll
        for (int i = 0; i < 2; ++i) {
            int idx = tid + i * 128;
            int r = idx >> 3, c8 = (idx & 7) << 3;
            union { uint4 u; bf16_t e[8]; } val;
            val.u = *(const uint4*)&Vg[base + (size_t)(k0 + r) * 1024 + c8];
            #pragma unroll
            for (int j = 0; j < 8; ++j) Vt[c8 + j][r] = val.e[j];
        }
        async_wait_le0();
        __syncthreads();

        // S(16x32) = Q * K^T
        #pragma unroll
        for (int nt = 0; nt < 2; ++nt) {
            v8f s = {};
            #pragma unroll
            for (int dt = 0; dt < 2; ++dt) {
                const bf16_t* row = &Kb[nt * 16 + l16][dt * 32];
                v16bf kf = join16(row + hf * 8, row + 16 + hf * 8);
                s = wmma_bf16(qf[dt], kf, s);
            }
            #pragma unroll
            for (int i = 0; i < 8; ++i)
                Ss[wave][hf * 8 + i][nt * 16 + l16] = s[i];
        }
        __syncthreads();

        // Online softmax: lanes 0..15 own one query row each
        if (lane < 16) {
            const int row = lane, q = q0w + row;
            float mo = mSt[wave][row];
            float mn = mo;
            #pragma unroll
            for (int k = 0; k < 32; ++k)
                if (k0 + k <= q) mn = fmaxf(mn, Ss[wave][row][k]);
            float cr = __expf(mo - mn);
            float sum = 0.f;
            #pragma unroll
            for (int k = 0; k < 32; ++k) {
                float p = (k0 + k <= q) ? __expf(Ss[wave][row][k] - mn) : 0.f;
                Pb[wave][row][k] = f2bf(p);
                sum += p;
            }
            mSt[wave][row] = mn;
            lSt[wave][row] = lSt[wave][row] * cr + sum;
            cSt[wave][row] = cr;
        }
        __syncthreads();

        // Rescale O, accumulate P(16x32) * V(32x64)
        float cr[8];
        #pragma unroll
        for (int i = 0; i < 8; ++i) cr[i] = cSt[wave][hf * 8 + i];
        const bf16_t* prow = &Pb[wave][l16][0];
        v16bf pf = join16(prow + hf * 8, prow + 16 + hf * 8);
        #pragma unroll
        for (int dt = 0; dt < 4; ++dt) {
            #pragma unroll
            for (int i = 0; i < 8; ++i) oacc[dt][i] *= cr[i];
            const bf16_t* vrow = &Vt[dt * 16 + l16][0];
            v16bf vf = join16(vrow + hf * 8, vrow + 16 + hf * 8);
            oacc[dt] = wmma_bf16(pf, vf, oacc[dt]);
        }
    }

    __syncthreads();
    float inv[8];
    #pragma unroll
    for (int i = 0; i < 8; ++i) inv[i] = 1.f / lSt[wave][hf * 8 + i];
    #pragma unroll
    for (int dt = 0; dt < 4; ++dt)
        #pragma unroll
        for (int i = 0; i < 8; ++i) {
            int m = q0w + hf * 8 + i;
            Og[base + (size_t)m * 1024 + dt * 16 + l16] = oacc[dt][i] * inv[i];
        }
}

// ---------------------------------------------------------------------------
// Fused residual + LayerNorm: out = LN(a + s*r)*g + beta, D = 1024.
// Optionally also emits a bf16 copy (for the following GEMM).
// ---------------------------------------------------------------------------
__global__ __launch_bounds__(256) void ln_residual(
    const float* __restrict__ a, const float* __restrict__ r,
    const float* __restrict__ g, const float* __restrict__ beta,
    float* __restrict__ out32, bf16_t* __restrict__ out16, float s)
{
    __shared__ float red[8];
    const int D = 1024;
    const int row = blockIdx.x;
    const int tid = threadIdx.x, wave = tid >> 5, lane = tid & 31;
    const float* ap = a + (size_t)row * D;
    const float* rp = r + (size_t)row * D;

    float vals[4];
    float sum = 0.f;
    #pragma unroll
    for (int i = 0; i < 4; ++i) {
        int c = tid + i * 256;
        vals[i] = ap[c] + s * rp[c];
        sum += vals[i];
    }
    #pragma unroll
    for (int off = 16; off > 0; off >>= 1) sum += __shfl_down(sum, off, 32);
    if (lane == 0) red[wave] = sum;
    __syncthreads();
    if (tid == 0) { float t = 0.f; for (int i = 0; i < 8; ++i) t += red[i]; red[0] = t; }
    __syncthreads();
    const float mean = red[0] * (1.f / D);

    float vs = 0.f;
    #pragma unroll
    for (int i = 0; i < 4; ++i) { float d = vals[i] - mean; vs += d * d; }
    __syncthreads();
    #pragma unroll
    for (int off = 16; off > 0; off >>= 1) vs += __shfl_down(vs, off, 32);
    if (lane == 0) red[wave] = vs;
    __syncthreads();
    if (tid == 0) { float t = 0.f; for (int i = 0; i < 8; ++i) t += red[i];
                    red[0] = rsqrtf(t * (1.f / D) + 1e-5f); }
    __syncthreads();
    const float rstd = red[0];

    #pragma unroll
    for (int i = 0; i < 4; ++i) {
        int c = tid + i * 256;
        float v = (vals[i] - mean) * rstd * g[c] + beta[c];
        out32[(size_t)row * D + c] = v;
        if (out16) out16[(size_t)row * D + c] = f2bf(v);
    }
}

// ---------------------------------------------------------------------------
// Host-side orchestration
// ---------------------------------------------------------------------------
extern "C" void kernel_launch(void* const* d_in, const int* in_sizes, int n_in,
                              void* d_out, int out_size, void* d_ws, size_t ws_size,
                              hipStream_t stream) {
    (void)in_sizes; (void)n_in; (void)out_size; (void)ws_size;
    const float* x   = (const float*)d_in[0];
    const float* Wq  = (const float*)d_in[2];
    const float* bq  = (const float*)d_in[3];
    const float* Wk  = (const float*)d_in[4];
    const float* bk  = (const float*)d_in[5];
    const float* Wv  = (const float*)d_in[6];
    const float* bv  = (const float*)d_in[7];
    const float* W1  = (const float*)d_in[8];
    const float* b1  = (const float*)d_in[9];
    const float* W2  = (const float*)d_in[10];
    const float* b2  = (const float*)d_in[11];
    const float* g1  = (const float*)d_in[12];
    const float* be1 = (const float*)d_in[13];
    const float* g2  = (const float*)d_in[14];
    const float* be2 = (const float*)d_in[15];
    float* out = (float*)d_out;

    const int Tv = 2048, Dm = 1024, Df = 4096;
    const int Mr = 2 * Tv;                   // 4096 rows
    const size_t MD = (size_t)Mr * Dm;       // 4M
    const size_t MF = (size_t)Mr * Df;       // 16M

    char* w = (char*)d_ws;
    float* attn32 = (float*)w; w += MD * 4;
    float* x1_32  = (float*)w; w += MD * 4;
    float* ff32   = (float*)w; w += MD * 4;
    bf16_t* xbf  = (bf16_t*)w; w += MD * 2;
    bf16_t* Qbf  = (bf16_t*)w; w += MD * 2;
    bf16_t* Kbf  = (bf16_t*)w; w += MD * 2;
    bf16_t* Vbf  = (bf16_t*)w; w += MD * 2;
    bf16_t* x1bf = (bf16_t*)w; w += MD * 2;
    bf16_t* WqT  = (bf16_t*)w; w += (size_t)Dm * Dm * 2;
    bf16_t* WkT  = (bf16_t*)w; w += (size_t)Dm * Dm * 2;
    bf16_t* WvT  = (bf16_t*)w; w += (size_t)Dm * Dm * 2;
    bf16_t* W1T  = (bf16_t*)w; w += (size_t)Dm * Df * 2;
    bf16_t* W2T  = (bf16_t*)w; w += (size_t)Df * Dm * 2;
    bf16_t* Hbf  = (bf16_t*)w; w += MF * 2;

    // Pre-pass: activations + weights to bf16 (weights transposed to [N][K])
    cvt_bf16<<<dim3((unsigned)(MD / 1024)), dim3(256), 0, stream>>>(x, xbf);
    transpose_cvt_bf16<<<dim3(Dm / 32, Dm / 32), dim3(256), 0, stream>>>(Wq, WqT, Dm, Dm);
    transpose_cvt_bf16<<<dim3(Dm / 32, Dm / 32), dim3(256), 0, stream>>>(Wk, WkT, Dm, Dm);
    transpose_cvt_bf16<<<dim3(Dm / 32, Dm / 32), dim3(256), 0, stream>>>(Wv, WvT, Dm, Dm);
    transpose_cvt_bf16<<<dim3(Df / 32, Dm / 32), dim3(256), 0, stream>>>(W1, W1T, Dm, Df);
    transpose_cvt_bf16<<<dim3(Dm / 32, Df / 32), dim3(256), 0, stream>>>(W2, W2T, Df, Dm);

    dim3 blk(256);
    // QKV projections (bf16 out, consumed by attention)
    gemm_bf16_wmma<<<dim3(Dm / 128, Mr / 128), blk, 0, stream>>>(xbf, WqT, bq, nullptr, Qbf, Mr, Dm, Dm, 0);
    gemm_bf16_wmma<<<dim3(Dm / 128, Mr / 128), blk, 0, stream>>>(xbf, WkT, bk, nullptr, Kbf, Mr, Dm, Dm, 0);
    gemm_bf16_wmma<<<dim3(Dm / 128, Mr / 128), blk, 0, stream>>>(xbf, WvT, bv, nullptr, Vbf, Mr, Dm, Dm, 0);
    // Causal flash attention
    attn_flash_wmma<<<dim3(Tv / 64, 16, 2), dim3(128), 0, stream>>>(Qbf, Kbf, Vbf, attn32, Tv);
    // x1 = LN(x + 2*attn), dual fp32 + bf16
    ln_residual<<<dim3(Mr), blk, 0, stream>>>(x, attn32, g1, be1, x1_32, x1bf, 2.0f);
    // FFN
    gemm_bf16_wmma<<<dim3(Df / 128, Mr / 128), blk, 0, stream>>>(x1bf, W1T, b1, nullptr, Hbf, Mr, Df, Dm, 1);
    gemm_bf16_wmma<<<dim3(Dm / 128, Mr / 128), blk, 0, stream>>>(Hbf, W2T, b2, ff32, nullptr, Mr, Dm, Df, 0);
    // out = LN(x1 + 2*ff)
    ln_residual<<<dim3(Mr), blk, 0, stream>>>(x1_32, ff32, g2, be2, out, nullptr, 2.0f);
}
